// Decoder_13271448945120
// MI455X (gfx1250) — matrix operand order repaired and checked by reading, not compiled, not run on hardware
//
#include <hip/hip_runtime.h>
#include <stdint.h>
#include <math.h>

// ---------------------------------------------------------------------------
// Types for CDNA5 WMMA (gfx1250, wave32): v_wmma_f32_16x16x32_bf16
// ---------------------------------------------------------------------------
typedef __attribute__((ext_vector_type(16))) __bf16 bf16x16;
typedef __attribute__((ext_vector_type(8)))  __bf16 bf16x8;
typedef __attribute__((ext_vector_type(8)))  float  v8f;

#define LOG2E 1.44269504088896340736f

__device__ __forceinline__ v8f zero_v8f() {
  v8f z = {0.f, 0.f, 0.f, 0.f, 0.f, 0.f, 0.f, 0.f};
  return z;
}

__device__ __forceinline__ v8f wmma_bf16(bf16x16 a, bf16x16 b, v8f c) {
  // (neg_a, A, neg_b, B, c_mod, C, reuse_a, reuse_b)
  return __builtin_amdgcn_wmma_f32_16x16x32_bf16(false, a, false, b, (short)0, c,
                                                 false, false);
}

// Async global->LDS copy, 16B per lane, tracked by ASYNCcnt (CDNA5 async
// path: no VGPR round trip for the data).
__device__ __forceinline__ void async_copy_b128(__bf16* lds_dst,
                                                const __bf16* gsrc) {
  const uint32_t laddr = (uint32_t)(uintptr_t)lds_dst;  // LDS byte offset
  asm volatile("global_load_async_to_lds_b128 %0, %1, off"
               :
               : "v"(laddr), "v"(gsrc)
               : "memory");
}

__device__ __forceinline__ void wait_asynccnt0() {
  asm volatile("s_wait_asynccnt 0x0" ::: "memory");
}

// A-matrix fragment (16x32 bf16, MxK). Per ISA 7.12.2:
// lane l: row = l&15, element e maps to K = (e&8)*2 + (l>>4)*8 + (e&7)
// -> two contiguous 8-element (16B) chunks at K0 = (l>>4)*8 and K0+16.
__device__ __forceinline__ bf16x16 load_a_frag(const __bf16* rowbase, int ld) {
  const int l = threadIdx.x & 31;
  const __bf16* p = rowbase + (size_t)(l & 15) * ld + ((l >> 4) * 8);
  const bf16x8 lo = *(const bf16x8*)(p);
  const bf16x8 hi = *(const bf16x8*)(p + 16);
  bf16x16 r;
#pragma unroll
  for (int i = 0; i < 8; i++) { r[i] = lo[i]; r[i + 8] = hi[i]; }
  return r;
}

// B-matrix fragment (32x16 bf16, KxN) from a K-contiguous ("transposed")
// layout: lane l -> col N = l&15, element e -> K = (e&8)*2 + (l>>4)*8 + (e&7).
__device__ __forceinline__ bf16x16 load_b_frag_contig(const __bf16* base, int ld) {
  const int l = threadIdx.x & 31;
  const __bf16* p = base + (size_t)(l & 15) * ld + ((l >> 4) * 8);
  const bf16x8 lo = *(const bf16x8*)(p);
  const bf16x8 hi = *(const bf16x8*)(p + 16);
  bf16x16 r;
#pragma unroll
  for (int i = 0; i < 8; i++) { r[i] = lo[i]; r[i + 8] = hi[i]; }
  return r;
}

// A fragment from an f32 16x32 LDS tile (convert to bf16 on the fly).
__device__ __forceinline__ bf16x16 load_p_frag(const float* pbase) {
  const int l = threadIdx.x & 31;
  const float* p = pbase + (l & 15) * 32 + ((l >> 4) * 8);
  bf16x16 r;
#pragma unroll
  for (int i = 0; i < 8; i++) {
    r[i] = (__bf16)p[i];
    r[i + 8] = (__bf16)p[i + 16];
  }
  return r;
}

// ---------------------------------------------------------------------------
// Elementwise f32 -> bf16 cast
// ---------------------------------------------------------------------------
__global__ void cast_f32_to_bf16_kernel(const float* __restrict__ in,
                                        __bf16* __restrict__ out, int n) {
  int i = blockIdx.x * blockDim.x + threadIdx.x;
  const int stride = gridDim.x * blockDim.x;
  for (; i < n; i += stride) out[i] = (__bf16)in[i];
}

// ---------------------------------------------------------------------------
// RMSNorm: one block per row, writes optional f32 and bf16 outputs
// ---------------------------------------------------------------------------
__global__ __launch_bounds__(256) void rmsnorm_kernel(
    const float* __restrict__ x, const float* __restrict__ g,
    float* __restrict__ yF, __bf16* __restrict__ yB, int D) {
  const int row = blockIdx.x;
  const float* xr = x + (size_t)row * D;
  float ss = 0.f;
  for (int i = threadIdx.x; i < D; i += 256) {
    const float v = xr[i];
    ss += v * v;
  }
  __shared__ float red[256];
  red[threadIdx.x] = ss;
  __syncthreads();
  for (int s = 128; s > 0; s >>= 1) {
    if (threadIdx.x < s) red[threadIdx.x] += red[threadIdx.x + s];
    __syncthreads();
  }
  const float inv = rsqrtf(red[0] / (float)D + 1e-6f);
  for (int i = threadIdx.x; i < D; i += 256) {
    const float v = xr[i] * inv * g[i];
    if (yF) yF[(size_t)row * D + i] = v;
    yB[(size_t)row * D + i] = (__bf16)v;
  }
}

// ---------------------------------------------------------------------------
// bf16 WMMA GEMM: C = alpha*(A@B) (+ resid). 128x128 tile, 8 waves,
// double-buffered software pipeline:
//   issue async A(t+1) + global B(t+1) loads -> 8 WMMAs on tile t
//   -> scatter B(t+1) to LDS -> s_wait_asynccnt 0 -> barrier.
// The async copy latency is hidden behind the WMMA pipe.
// ---------------------------------------------------------------------------
__global__ __launch_bounds__(256) void gemm_bf16_kernel(
    const __bf16* __restrict__ A, const __bf16* __restrict__ Bw,
    int Mm, int Nn, int Kk, float alpha,
    const float* __restrict__ resid, float* __restrict__ outF,
    __bf16* __restrict__ outB) {
  __shared__ __align__(64) __bf16 As[2][128][32];   // 16KB
  __shared__ __align__(64) __bf16 BsT[2][128][32];  // 16KB, [n][k]
  const int tid = threadIdx.x;
  const int l = tid & 31;
  const int w = tid >> 5;
  const int mBase = blockIdx.y * 128;
  const int nBase = blockIdx.x * 128;
  const int mW = (w >> 1) * 32;
  const int nW = (w & 1) * 64;
  v8f acc[2][4];
#pragma unroll
  for (int r = 0; r < 2; r++)
#pragma unroll
    for (int t = 0; t < 4; t++) acc[r][t] = zero_v8f();

  const int nTiles = Kk >> 5;

  // ---- stage tile 0 ----
#pragma unroll
  for (int i = 0; i < 2; i++) {
    const int lin = (tid + i * 256) * 8;
    const int r = lin >> 5, c = lin & 31;
    async_copy_b128(&As[0][r][c], &A[(size_t)(mBase + r) * Kk + c]);
  }
#pragma unroll
  for (int i = 0; i < 2; i++) {
    const int lin = (tid + i * 256) * 8;
    const int kk = lin >> 7, n = lin & 127;
    const bf16x8 vb = *(const bf16x8*)&Bw[(size_t)kk * Nn + nBase + n];
#pragma unroll
    for (int j = 0; j < 8; j++) BsT[0][n + j][kk] = vb[j];
  }
  wait_asynccnt0();
  __syncthreads();

  for (int t = 0; t < nTiles; t++) {
    const int cur = t & 1;
    const int nxt = cur ^ 1;
    const int kb2 = (t + 1) << 5;
    const bool hasNext = (t + 1) < nTiles;
    bf16x8 bv[2];
    if (hasNext) {  // prefetch tile t+1: A via async, B into VGPRs
#pragma unroll
      for (int i = 0; i < 2; i++) {
        const int lin = (tid + i * 256) * 8;
        const int r = lin >> 5, c = lin & 31;
        async_copy_b128(&As[nxt][r][c],
                        &A[(size_t)(mBase + r) * Kk + kb2 + c]);
      }
#pragma unroll
      for (int i = 0; i < 2; i++) {
        const int lin = (tid + i * 256) * 8;
        const int kk = lin >> 7, n = lin & 127;
        bv[i] = *(const bf16x8*)&Bw[(size_t)(kb2 + kk) * Nn + nBase + n];
      }
    }
    // compute on tile t (hides prefetch latency)
    const bf16x16 a0 = load_a_frag(&As[cur][mW][0], 32);
    const bf16x16 a1 = load_a_frag(&As[cur][mW + 16][0], 32);
#pragma unroll
    for (int tt = 0; tt < 4; tt++) {
      const bf16x16 bt = load_b_frag_contig(&BsT[cur][nW + tt * 16][0], 32);
      acc[0][tt] = wmma_bf16(a0, bt, acc[0][tt]);
      acc[1][tt] = wmma_bf16(a1, bt, acc[1][tt]);
    }
    if (hasNext) {  // scatter B(t+1) into the other LDS buffer
#pragma unroll
      for (int i = 0; i < 2; i++) {
        const int lin = (tid + i * 256) * 8;
        const int kk = lin >> 7, n = lin & 127;
#pragma unroll
        for (int j = 0; j < 8; j++) BsT[nxt][n + j][kk] = bv[i][j];
      }
      wait_asynccnt0();
    }
    __syncthreads();
  }

  const int half = l >> 4, lane16 = l & 15;
#pragma unroll
  for (int r = 0; r < 2; r++)
#pragma unroll
    for (int t = 0; t < 4; t++)
#pragma unroll
      for (int v = 0; v < 8; v++) acc[r][t][v] *= alpha;
  if (resid) {
#pragma unroll
    for (int r = 0; r < 2; r++)
#pragma unroll
      for (int t = 0; t < 4; t++) {
        const int row0 = mBase + mW + r * 16 + half * 8;
        const int col = nBase + nW + t * 16 + lane16;
        size_t idx = (size_t)row0 * Nn + col;
#pragma unroll
        for (int v = 0; v < 8; v++, idx += (size_t)Nn)
          acc[r][t][v] += resid[idx];
      }
  }
  if (outF) {
#pragma unroll
    for (int r = 0; r < 2; r++)
#pragma unroll
      for (int t = 0; t < 4; t++) {
        const int row0 = mBase + mW + r * 16 + half * 8;
        const int col = nBase + nW + t * 16 + lane16;
        size_t idx = (size_t)row0 * Nn + col;
#pragma unroll
        for (int v = 0; v < 8; v++, idx += (size_t)Nn) outF[idx] = acc[r][t][v];
      }
  }
  if (outB) {
#pragma unroll
    for (int r = 0; r < 2; r++)
#pragma unroll
      for (int t = 0; t < 4; t++) {
        const int row0 = mBase + mW + r * 16 + half * 8;
        const int col = nBase + nW + t * 16 + lane16;
        size_t idx = (size_t)row0 * Nn + col;
#pragma unroll
        for (int v = 0; v < 8; v++, idx += (size_t)Nn)
          outB[idx] = (__bf16)acc[r][t][v];
      }
  }
}

// ---------------------------------------------------------------------------
// Fused GEMM + SwiGLU: u = silu(A@W1) * (A@W3), bf16 out. 128x64 tile,
// 8 waves, same double-buffered pipeline as gemm_bf16_kernel.
// ---------------------------------------------------------------------------
__global__ __launch_bounds__(256) void gemm_swiglu_kernel(
    const __bf16* __restrict__ A, const __bf16* __restrict__ W1,
    const __bf16* __restrict__ W3, int Mm, int Nn, int Kk,
    __bf16* __restrict__ outB) {
  __shared__ __align__(64) __bf16 As[2][128][32];  // 16KB
  __shared__ __align__(64) __bf16 B1T[2][64][32];  // 8KB
  __shared__ __align__(64) __bf16 B3T[2][64][32];  // 8KB
  const int tid = threadIdx.x;
  const int l = tid & 31;
  const int w = tid >> 5;
  const int mBase = blockIdx.y * 128;
  const int nBase = blockIdx.x * 64;
  const int mW = (w >> 1) * 32;
  const int nW = (w & 1) * 32;
  v8f acc1[2][2], acc3[2][2];
#pragma unroll
  for (int r = 0; r < 2; r++)
#pragma unroll
    for (int t = 0; t < 2; t++) {
      acc1[r][t] = zero_v8f();
      acc3[r][t] = zero_v8f();
    }

  const int nTiles = Kk >> 5;
  const int blin = tid * 8;
  const int bkk = blin >> 6, bn = blin & 63;

  // ---- stage tile 0 ----
#pragma unroll
  for (int i = 0; i < 2; i++) {
    const int lin = (tid + i * 256) * 8;
    const int r = lin >> 5, c = lin & 31;
    async_copy_b128(&As[0][r][c], &A[(size_t)(mBase + r) * Kk + c]);
  }
  {
    const bf16x8 v1 = *(const bf16x8*)&W1[(size_t)bkk * Nn + nBase + bn];
    const bf16x8 v3 = *(const bf16x8*)&W3[(size_t)bkk * Nn + nBase + bn];
#pragma unroll
    for (int j = 0; j < 8; j++) {
      B1T[0][bn + j][bkk] = v1[j];
      B3T[0][bn + j][bkk] = v3[j];
    }
  }
  wait_asynccnt0();
  __syncthreads();

  for (int t = 0; t < nTiles; t++) {
    const int cur = t & 1;
    const int nxt = cur ^ 1;
    const int kb2 = (t + 1) << 5;
    const bool hasNext = (t + 1) < nTiles;
    bf16x8 bv1, bv3;
    if (hasNext) {
#pragma unroll
      for (int i = 0; i < 2; i++) {
        const int lin = (tid + i * 256) * 8;
        const int r = lin >> 5, c = lin & 31;
        async_copy_b128(&As[nxt][r][c],
                        &A[(size_t)(mBase + r) * Kk + kb2 + c]);
      }
      bv1 = *(const bf16x8*)&W1[(size_t)(kb2 + bkk) * Nn + nBase + bn];
      bv3 = *(const bf16x8*)&W3[(size_t)(kb2 + bkk) * Nn + nBase + bn];
    }
    const bf16x16 a0 = load_a_frag(&As[cur][mW][0], 32);
    const bf16x16 a1 = load_a_frag(&As[cur][mW + 16][0], 32);
#pragma unroll
    for (int tt = 0; tt < 2; tt++) {
      const bf16x16 b1t = load_b_frag_contig(&B1T[cur][nW + tt * 16][0], 32);
      const bf16x16 b3t = load_b_frag_contig(&B3T[cur][nW + tt * 16][0], 32);
      acc1[0][tt] = wmma_bf16(a0, b1t, acc1[0][tt]);
      acc1[1][tt] = wmma_bf16(a1, b1t, acc1[1][tt]);
      acc3[0][tt] = wmma_bf16(a0, b3t, acc3[0][tt]);
      acc3[1][tt] = wmma_bf16(a1, b3t, acc3[1][tt]);
    }
    if (hasNext) {
#pragma unroll
      for (int j = 0; j < 8; j++) {
        B1T[nxt][bn + j][bkk] = bv1[j];
        B3T[nxt][bn + j][bkk] = bv3[j];
      }
      wait_asynccnt0();
    }
    __syncthreads();
  }

  const int half = l >> 4, lane16 = l & 15;
#pragma unroll
  for (int r = 0; r < 2; r++)
#pragma unroll
    for (int t = 0; t < 2; t++) {
      const int row0 = mBase + mW + r * 16 + half * 8;
      const int col = nBase + nW + t * 16 + lane16;
      size_t idx = (size_t)row0 * Nn + col;
#pragma unroll
      for (int v = 0; v < 8; v++, idx += (size_t)Nn) {
        const float a = acc1[r][t][v];
        const float g = a / (1.f + __expf(-a));  // silu
        outB[idx] = (__bf16)(g * acc3[r][t][v]);
      }
    }
}

// ---------------------------------------------------------------------------
// Differential flash attention. Block = 64 threads = 2 waves.
// Wave w handles softmax component w (Q_w K_w^T); V tile is shared between
// components and staged cooperatively into LDS (transposed); combine
// O0 - lam*O1 through LDS at the end.
// Q,K: [B*S, H*2*HS] (col = h*128 + c*64 + d). V: [B*S, H*HS]. Out bf16.
// ---------------------------------------------------------------------------
__global__ __launch_bounds__(64) void diff_flash_attn_kernel(
    const __bf16* __restrict__ Q, const __bf16* __restrict__ Km,
    const __bf16* __restrict__ V, const int* __restrict__ pad,
    const float* __restrict__ lam, __bf16* __restrict__ Out,
    int S, int causal) {
  __shared__ __align__(64) float Pb[2][16][32];   // per-wave P staging (4KB)
  __shared__ __align__(64) float Ob[2][16][64];   // per-wave O staging (8KB)
  __shared__ __align__(64) __bf16 VsT[64][32];    // shared V tile [d][k] (4KB)
  const int tid = threadIdx.x;
  const int l = tid & 31;
  const int w = tid >> 5;  // differential component 0/1
  const int qBase = blockIdx.x * 16;
  const int h = blockIdx.y;
  const int b = blockIdx.z;
  const int lane16 = l & 15;
  const int half = l >> 4;

  const __bf16* Qbase =
      Q + ((size_t)(b * S + qBase)) * 2048 + h * 128 + w * 64;
  const bf16x16 qf0 = load_a_frag(Qbase, 2048);       // K-dim 0..31
  const bf16x16 qf1 = load_a_frag(Qbase + 32, 2048);  // K-dim 32..63

  float mrow[8], srow[8];
  v8f o[4];
#pragma unroll
  for (int v = 0; v < 8; v++) { mrow[v] = -1e30f; srow[v] = 0.f; }
#pragma unroll
  for (int t = 0; t < 4; t++) o[t] = zero_v8f();

  const int kmax = causal ? (qBase + 16) : S;
  for (int kb = 0; kb < kmax; kb += 32) {
    // Cooperative V tile stage: 32 keys x 64 dims -> VsT[d][k]
#pragma unroll
    for (int i = 0; i < 4; i++) {
      const int lin = (tid + i * 64) * 8;
      const int kk = lin >> 6, dd = lin & 63;
      const bf16x8 vv =
          *(const bf16x8*)&V[((size_t)(b * S + kb + kk)) * 1024 + h * 64 + dd];
#pragma unroll
      for (int j = 0; j < 8; j++) VsT[dd + j][kk] = vv[j];
    }
    __syncthreads();

    const __bf16* Kbase =
        Km + ((size_t)(b * S + kb)) * 2048 + h * 128 + w * 64;
    v8f s0 = zero_v8f(), s1 = zero_v8f();
    {
      const bf16x16 k00 = load_b_frag_contig(Kbase, 2048);
      const bf16x16 k01 = load_b_frag_contig(Kbase + 32, 2048);
      s0 = wmma_bf16(qf0, k00, s0);
      s0 = wmma_bf16(qf1, k01, s0);
      const bf16x16 k10 = load_b_frag_contig(Kbase + (size_t)16 * 2048, 2048);
      const bf16x16 k11 =
          load_b_frag_contig(Kbase + (size_t)16 * 2048 + 32, 2048);
      s1 = wmma_bf16(qf0, k10, s1);
      s1 = wmma_bf16(qf1, k11, s1);
    }
    const int col0 = kb + lane16;
    const int col1 = kb + 16 + lane16;
    const bool p0ok = pad[b * S + col0] != 0;
    const bool p1ok = pad[b * S + col1] != 0;
#pragma unroll
    for (int v = 0; v < 8; v++) {
      const int row = qBase + half * 8 + v;
      float x0 = (p0ok && (!causal || col0 <= row)) ? s0[v] * 0.125f : -1e9f;
      float x1 = (p1ok && (!causal || col1 <= row)) ? s1[v] * 0.125f : -1e9f;
      float lm = fmaxf(x0, x1);
#pragma unroll
      for (int off = 1; off < 16; off <<= 1)
        lm = fmaxf(lm, __shfl_xor(lm, off, 32));
      const float nm = fmaxf(mrow[v], lm);
      const float corr = exp2f((mrow[v] - nm) * LOG2E);
      const float p0 = exp2f((x0 - nm) * LOG2E);
      const float p1 = exp2f((x1 - nm) * LOG2E);
      mrow[v] = nm;
      float rs = p0 + p1;
#pragma unroll
      for (int off = 1; off < 16; off <<= 1) rs += __shfl_xor(rs, off, 32);
      srow[v] = srow[v] * corr + rs;
#pragma unroll
      for (int t = 0; t < 4; t++) o[t][v] *= corr;
      Pb[w][half * 8 + v][lane16] = p0;
      Pb[w][half * 8 + v][16 + lane16] = p1;
    }
    // wave-internal LDS visibility before re-reading P in A-fragment layout
    asm volatile("s_wait_dscnt 0x0" ::: "memory");
    const bf16x16 pf = load_p_frag(&Pb[w][0][0]);
#pragma unroll
    for (int t = 0; t < 4; t++) {
      const bf16x16 vf = load_b_frag_contig(&VsT[t * 16][0], 32);
      o[t] = wmma_bf16(pf, vf, o[t]);
    }
    __syncthreads();  // protect VsT before next tile's staging
  }

#pragma unroll
  for (int v = 0; v < 8; v++) {
    const float inv = 1.0f / srow[v];
    const int r = half * 8 + v;
#pragma unroll
    for (int t = 0; t < 4; t++) Ob[w][r][t * 16 + lane16] = o[t][v] * inv;
  }
  __syncthreads();
  const float lamh = lam[h];
  for (int i = tid; i < 16 * 64; i += 64) {
    const int r = i >> 6, d = i & 63;
    const float val = Ob[0][r][d] - lamh * Ob[1][r][d];
    Out[((size_t)(b * S + qBase + r)) * 1024 + h * 64 + d] = (__bf16)val;
  }
}

// ---------------------------------------------------------------------------
// Host orchestration
// ---------------------------------------------------------------------------
extern "C" void kernel_launch(void* const* d_in, const int* in_sizes, int n_in,
                              void* d_out, int out_size, void* d_ws,
                              size_t ws_size, hipStream_t stream) {
  (void)in_sizes; (void)n_in; (void)out_size; (void)ws_size;
  const int Bb = 4, S = 1024, D = 1024, H = 16, HS = 64, FF = 4096;
  const int Mr = Bb * S;  // 4096 rows
  (void)HS;

  const float* x      = (const float*)d_in[0];
  const float* enc    = (const float*)d_in[1];
  const int*   pad    = (const int*)d_in[2];
  const float* sa_wq  = (const float*)d_in[3];
  const float* sa_wk  = (const float*)d_in[4];
  const float* sa_wv  = (const float*)d_in[5];
  const float* sa_wo  = (const float*)d_in[6];
  const float* sa_lam = (const float*)d_in[7];
  const float* ca_wq  = (const float*)d_in[8];
  const float* ca_wk  = (const float*)d_in[9];
  const float* ca_wv  = (const float*)d_in[10];
  const float* ca_wo  = (const float*)d_in[11];
  const float* ca_lam = (const float*)d_in[12];
  const float* g1     = (const float*)d_in[13];
  const float* g2     = (const float*)d_in[14];
  const float* w1     = (const float*)d_in[15];
  const float* w2     = (const float*)d_in[16];
  const float* w3     = (const float*)d_in[17];

  char* ws = (char*)d_ws;
  size_t off = 0;
  auto abf = [&](size_t n) {
    __bf16* p = (__bf16*)(ws + off);
    off += (n * sizeof(__bf16) + 255) & ~(size_t)255;
    return p;
  };
  auto af32 = [&](size_t n) {
    float* p = (float*)(ws + off);
    off += (n * sizeof(float) + 255) & ~(size_t)255;
    return p;
  };

  __bf16* sa_wq_b = abf((size_t)D * 2048);
  __bf16* sa_wk_b = abf((size_t)D * 2048);
  __bf16* sa_wv_b = abf((size_t)D * 1024);
  __bf16* sa_wo_b = abf((size_t)1024 * D);
  __bf16* ca_wq_b = abf((size_t)D * 2048);
  __bf16* ca_wk_b = abf((size_t)D * 2048);
  __bf16* ca_wv_b = abf((size_t)D * 1024);
  __bf16* ca_wo_b = abf((size_t)1024 * D);
  __bf16* w1_b    = abf((size_t)D * FF);
  __bf16* w2_b    = abf((size_t)FF * D);
  __bf16* w3_b    = abf((size_t)D * FF);
  __bf16* enc_b   = abf((size_t)Mr * D);
  __bf16* xn1_b   = abf((size_t)Mr * D);
  __bf16* h1_b    = abf((size_t)Mr * D);
  __bf16* xn2_b   = abf((size_t)Mr * D);
  __bf16* attn_b  = abf((size_t)Mr * 1024);
  __bf16* Qb      = abf((size_t)Mr * 2048);
  __bf16* Kb      = abf((size_t)Mr * 2048);  // contiguous with Qb
  __bf16* Vb      = abf((size_t)Mr * 1024);
  float*  xn1_f   = af32((size_t)Mr * D);
  float*  h2_f    = af32((size_t)Mr * D);
  // SwiGLU intermediate (Mr x FF bf16 = 32MB) reuses the retired Q+K region.
  __bf16* u_b = Qb;

  auto cast = [&](const float* src, __bf16* dst, size_t n) {
    cast_f32_to_bf16_kernel<<<dim3(1024), dim3(256), 0, stream>>>(src, dst,
                                                                  (int)n);
  };
  auto gemm = [&](const __bf16* A, const __bf16* Bw, int Mm, int Nn, int Kk,
                  float alpha, const float* resid, float* outF, __bf16* outB) {
    gemm_bf16_kernel<<<dim3(Nn / 128, Mm / 128), dim3(256), 0, stream>>>(
        A, Bw, Mm, Nn, Kk, alpha, resid, outF, outB);
  };

  // weight / encoder conversions
  cast(enc, enc_b, (size_t)Mr * D);
  cast(sa_wq, sa_wq_b, (size_t)D * 2048);
  cast(sa_wk, sa_wk_b, (size_t)D * 2048);
  cast(sa_wv, sa_wv_b, (size_t)D * 1024);
  cast(sa_wo, sa_wo_b, (size_t)1024 * D);
  cast(ca_wq, ca_wq_b, (size_t)D * 2048);
  cast(ca_wk, ca_wk_b, (size_t)D * 2048);
  cast(ca_wv, ca_wv_b, (size_t)D * 1024);
  cast(ca_wo, ca_wo_b, (size_t)1024 * D);
  cast(w1, w1_b, (size_t)D * FF);
  cast(w2, w2_b, (size_t)FF * D);
  cast(w3, w3_b, (size_t)D * FF);

  // xn1 = rmsnorm(x, g1)
  rmsnorm_kernel<<<dim3(Mr), dim3(256), 0, stream>>>(x, g1, xn1_f, xn1_b, D);

  // self-attention Q/K/V projections
  gemm(xn1_b, sa_wq_b, Mr, 2048, 1024, 1.f, nullptr, nullptr, Qb);
  gemm(xn1_b, sa_wk_b, Mr, 2048, 1024, 1.f, nullptr, nullptr, Kb);
  gemm(xn1_b, sa_wv_b, Mr, 1024, 1024, 1.f, nullptr, nullptr, Vb);

  // causal differential flash attention
  diff_flash_attn_kernel<<<dim3(S / 16, H, Bb), dim3(64), 0, stream>>>(
      Qb, Kb, Vb, pad, sa_lam, attn_b, S, 1);

  // h1 = attn @ wo + xn1 (bf16 only; h1 feeds cross-attn Q, no later residual)
  gemm(attn_b, sa_wo_b, Mr, 1024, 1024, 1.f, xn1_f, nullptr, h1_b);

  // cross-attention projections (Q from h1, K/V from encoder output)
  gemm(h1_b, ca_wq_b, Mr, 2048, 1024, 1.f, nullptr, nullptr, Qb);
  gemm(enc_b, ca_wk_b, Mr, 2048, 1024, 1.f, nullptr, nullptr, Kb);
  gemm(enc_b, ca_wv_b, Mr, 1024, 1024, 1.f, nullptr, nullptr, Vb);

  // non-causal differential flash attention
  diff_flash_attn_kernel<<<dim3(S / 16, H, Bb), dim3(64), 0, stream>>>(
      Qb, Kb, Vb, pad, ca_lam, attn_b, S, 0);

  // h2 = c + c = 2 * (attn @ wo)
  gemm(attn_b, ca_wo_b, Mr, 1024, 1024, 2.f, nullptr, h2_f, nullptr);

  // xn2 = rmsnorm(h2, g2)
  rmsnorm_kernel<<<dim3(Mr), dim3(256), 0, stream>>>(h2_f, g2, nullptr, xn2_b,
                                                     D);

  // u = silu(xn2 @ w1) * (xn2 @ w3)
  gemm_swiglu_kernel<<<dim3(FF / 64, Mr / 128), dim3(256), 0, stream>>>(
      xn2_b, w1_b, w3_b, Mr, FF, 1024, u_b);

  // out = u @ w2 + h2
  gemm(u_b, w2_b, Mr, 1024, FF, 1.f, h2_f, (float*)d_out, nullptr);
}
